// QLSTM__gen041_65481071396870
// MI455X (gfx1250) — matrix operand order repaired and checked by reading, not compiled, not run on hardware
//
#include <hip/hip_runtime.h>
#include <hip/hip_bf16.h>

// ---------------------------------------------------------------------------
// Problem constants
// ---------------------------------------------------------------------------
#define QT  256      // timesteps
#define QB  1024     // batch
#define QD  128      // input dim
#define QH  4        // hidden (= n_qubits)
#define QW  132      // D + H  (weight row length)
#define NGQ 16       // 4 gates * 4 qubits

typedef __attribute__((ext_vector_type(16))) _Float16 v16h;
typedef __attribute__((ext_vector_type(8)))  float    v8f;

__device__ __forceinline__ float qsigmoid(float x) {
    return 1.0f / (1.0f + __expf(-x));
}

// ---------------------------------------------------------------------------
// Kernel 1: xdot[t*B+b, g*4+q] = x[t,b,:128] . W_gate[q,:128]
// GEMM M=T*B=262144, N=16, K=128 via v_wmma_f32_16x16x32_f16.
// One wave per 16-row tile. A/B/C register layouts per CDNA5 ISA 7.12.2.
// ---------------------------------------------------------------------------
__global__ __launch_bounds__(256) void qlstm_gemm_wmma(
    const float* __restrict__ x,
    const float* __restrict__ Wf, const float* __restrict__ Wi,
    const float* __restrict__ Wg, const float* __restrict__ Wo,
    float* __restrict__ xdot)
{
    const int lane = threadIdx.x & 31;
    const int wave = threadIdx.x >> 5;
    const int tile = blockIdx.x * 8 + wave;     // 16384 tiles of 16 rows
    const int half = lane >> 4;                 // 0 | 1 (lane half)
    const int nloc = lane & 15;                 // N column / M row-in-tile

    // ---- B matrix (weights, K x N = 128 x 16), 4 chunks of 32xK each -----
    // Lane column n = g*4 + q. Elements 0..7 hold K = c*32 + half*8 + j,
    // elements 8..15 hold K = c*32 + 16 + half*8 + j (2 packed f16 per VGPR).
    const float* Wptr = (nloc < 4) ? Wf : (nloc < 8) ? Wi : (nloc < 12) ? Wg : Wo;
    const float* wrow = Wptr + (size_t)(nloc & 3) * QW;   // row q, 132 floats
    v16h bmat[4];
    #pragma unroll
    for (int c = 0; c < 4; ++c) {
        const int k0 = c * 32 + half * 8;
        #pragma unroll
        for (int e = 0; e < 16; ++e)
            bmat[c][e] = (_Float16)wrow[k0 + (e & 7) + (e >> 3) * 16];
    }

    // ---- A matrix (activations, 16 rows x 128 K), 4 chunks of 16x32 ------
    // Lane row m = nloc. Elements 0..7 hold K = c*32 + half*8 + j,
    // elements 8..15 hold K = c*32 + 16 + half*8 + j.
    const float* arow = x + (size_t)(tile * 16 + nloc) * QD;
    v16h amat[4];
    #pragma unroll
    for (int c = 0; c < 4; ++c) {
        const int ka = c * 32 + half * 8;
        #pragma unroll
        for (int j = 0; j < 8; ++j) {
            amat[c][j]     = (_Float16)arow[ka + j];
            amat[c][8 + j] = (_Float16)arow[ka + 16 + j];
        }
    }

    // ---- 4 x v_wmma_f32_16x16x32_f16, f32 accumulate ---------------------
    v8f acc = {};
    #pragma unroll
    for (int c = 0; c < 4; ++c)
        acc = __builtin_amdgcn_wmma_f32_16x16x32_f16(
            /*neg_a=*/false, amat[c], /*neg_b=*/false, bmat[c],
            /*c_mod=*/(short)0, acc, /*reuse_a=*/false, /*reuse_b=*/false);

    // ---- store C: lane holds rows r + 8*half, col nloc -------------------
    float* orow = xdot + (size_t)tile * 16 * NGQ;
    #pragma unroll
    for (int r = 0; r < 8; ++r)
        orow[(size_t)(r + 8 * half) * NGQ + nloc] = acc[r];
}

// ---------------------------------------------------------------------------
// Kernel 2: the strictly-serial recurrence (256 steps). One block, 1024 thr.
// hx/cx collapse to [4] vectors; qlayer has closed form via cosine products.
// ---------------------------------------------------------------------------
__global__ __launch_bounds__(1024) void qlstm_seq(
    const float* __restrict__ xdot,
    const float* __restrict__ Wf, const float* __restrict__ bf_,
    const float* __restrict__ Wi, const float* __restrict__ bi_,
    const float* __restrict__ Wg, const float* __restrict__ bg_,
    const float* __restrict__ Wo, const float* __restrict__ bo_,
    const float* __restrict__ fclw, const float* __restrict__ fclb,
    const float* __restrict__ rxf, const float* __restrict__ rxi,
    const float* __restrict__ rxg, const float* __restrict__ rxo,
    float* __restrict__ hseq,   // [256][4]
    float* __restrict__ cfin)   // [4]
{
    __shared__ float s_hx[QH];
    __shared__ float s_cx[QH];
    __shared__ float s_ht[NGQ];        // hterm[g*4+q]
    __shared__ float s_red[4][32];     // per-gate per-wave partials

    const int tid  = threadIdx.x;
    const int lane = tid & 31;
    const int wv   = tid >> 5;
    const float w  = fclw[0];
    const float bb = fclb[0];

    if (tid < QH) { s_hx[tid] = 0.0f; s_cx[tid] = 0.0f; }
    __syncthreads();

    for (int t = 0; t < QT; ++t) {
        // recurrent term: hterm[g,q] = b_g[q] + sum_h hx[h]*W_g[q,128+h]
        if (tid < NGQ) {
            const int g = tid >> 2, q = tid & 3;
            const float* Wr = (g == 0 ? Wf : g == 1 ? Wi : g == 2 ? Wg : Wo)
                              + (size_t)q * QW + QD;
            const float* br = (g == 0 ? bf_ : g == 1 ? bi_ : g == 2 ? bg_ : bo_);
            float s = br[q];
            #pragma unroll
            for (int h = 0; h < QH; ++h) s += s_hx[h] * Wr[h];
            s_ht[tid] = s;
        }
        __syncthreads();

        // each thread owns batch row b = tid : 16 tanh partials -> 4 gate sums
        const float* row = xdot + ((size_t)t * QB + tid) * NGQ;
        float p[4];
        #pragma unroll
        for (int g = 0; g < 4; ++g) {
            float s = 0.0f;
            #pragma unroll
            for (int q = 0; q < 4; ++q) {
                const float v = row[g * 4 + q] + s_ht[g * 4 + q];
                s += tanhf(v * w + bb);
            }
            p[g] = s;
        }
        #pragma unroll
        for (int g = 0; g < 4; ++g)
            #pragma unroll
            for (int off = 16; off > 0; off >>= 1)
                p[g] += __shfl_down(p[g], off, 32);
        if (lane == 0) {
            #pragma unroll
            for (int g = 0; g < 4; ++g) s_red[g][wv] = p[g];
        }
        __syncthreads();

        if (tid == 0) {
            float theta[4];
            #pragma unroll
            for (int g = 0; g < 4; ++g) {
                float s = 0.0f;
                for (int j = 0; j < 32; ++j) s += s_red[g][j];
                theta[g] = s * (1.0f / (QB * 4.0f));   // mean over B*NQ
            }
            // qlayer closed form: ev = products of cos(theta + rx_j)
            const float* rx[4] = {rxf, rxi, rxg, rxo};
            float ev[4][4];
            #pragma unroll
            for (int g = 0; g < 4; ++g) {
                float c0 = __cosf(theta[g] + rx[g][0]);
                float c1 = __cosf(theta[g] + rx[g][1]);
                float c2 = __cosf(theta[g] + rx[g][2]);
                float c3 = __cosf(theta[g] + rx[g][3]);
                ev[g][0] = c1 * c2 * c3;       // Z0 <- b1^b2^b3
                ev[g][1] = c0 * c1;            // Z1 <- b0^b1
                ev[g][2] = c0 * c1 * c2;       // Z2 <- b0^b1^b2
                ev[g][3] = c0 * c1 * c2 * c3;  // Z3 <- b0^b1^b2^b3
            }
            #pragma unroll
            for (int q = 0; q < QH; ++q) {
                const float fv = qsigmoid(ev[0][q]);
                const float iv = qsigmoid(ev[1][q]);
                const float gv = tanhf(ev[2][q]);
                const float ov = qsigmoid(ev[3][q]);
                const float c2 = fv * s_cx[q] + iv * gv;
                const float h2 = ov * tanhf(c2);
                s_cx[q] = c2;
                s_hx[q] = h2;
                hseq[t * QH + q] = h2;
                if (t == QT - 1) cfin[q] = c2;
            }
        }
        __syncthreads();
    }
}

// ---------------------------------------------------------------------------
// Kernel 3: broadcast-writer. outputs[t,b,h]=hseq[t,h]; hx=hseq[T-1]; cx=cfin.
// ---------------------------------------------------------------------------
__global__ __launch_bounds__(256) void qlstm_out(
    const float* __restrict__ hseq, const float* __restrict__ cfin,
    float* __restrict__ out)
{
    const size_t TBH = (size_t)QT * QB * QH;              // 1,048,576
    const size_t idx = (size_t)blockIdx.x * 256 + threadIdx.x;
    const int h = (int)(idx & 3);
    if (idx < TBH) {
        const size_t t = idx >> 12;                       // / (B*H)
        out[idx] = hseq[t * QH + h];
    } else if (idx < TBH + (size_t)QB * QH) {             // final hx
        out[idx] = hseq[(QT - 1) * QH + h];
    } else if (idx < TBH + 2 * (size_t)QB * QH) {         // final cx
        out[idx] = cfin[h];
    }
}

// ---------------------------------------------------------------------------
extern "C" void kernel_launch(void* const* d_in, const int* in_sizes, int n_in,
                              void* d_out, int out_size, void* d_ws, size_t ws_size,
                              hipStream_t stream) {
    const float* x    = (const float*)d_in[0];
    const float* Wf   = (const float*)d_in[1];
    const float* bf_  = (const float*)d_in[2];
    const float* Wi   = (const float*)d_in[3];
    const float* bi_  = (const float*)d_in[4];
    const float* Wg   = (const float*)d_in[5];
    const float* bg_  = (const float*)d_in[6];
    const float* Wo   = (const float*)d_in[7];
    const float* bo_  = (const float*)d_in[8];
    const float* fclw = (const float*)d_in[9];
    const float* fclb = (const float*)d_in[10];
    const float* rxf  = (const float*)d_in[11];
    const float* rxi  = (const float*)d_in[12];
    const float* rxg  = (const float*)d_in[13];
    const float* rxo  = (const float*)d_in[14];

    // workspace layout: xdot [T*B, 16] f32 (16 MB) | hseq [256][4] | cfin [4]
    float* xdot = (float*)d_ws;
    float* hseq = (float*)((char*)d_ws + (size_t)QT * QB * NGQ * sizeof(float));
    float* cfin = hseq + QT * QH;

    // Phase A: WMMA GEMM, 16384 row-tiles, 8 waves per 256-thread block.
    qlstm_gemm_wmma<<<dim3(2048), dim3(256), 0, stream>>>(x, Wf, Wi, Wg, Wo, xdot);

    // Phase B: serial 256-step recurrence on a single WGP (xdot L2-resident).
    qlstm_seq<<<dim3(1), dim3(1024), 0, stream>>>(
        xdot, Wf, bf_, Wi, bi_, Wg, bg_, Wo, bo_,
        fclw, fclb, rxf, rxi, rxg, rxo, hseq, cfin);

    // Phase C: broadcast results into d_out (outputs, hx, cx concatenated).
    const size_t total = (size_t)QT * QB * QH + 2 * (size_t)QB * QH;
    qlstm_out<<<dim3((unsigned)((total + 255) / 256)), dim3(256), 0, stream>>>(
        hseq, cfin, (float*)d_out);
}
